// CrossAttn_22050362098064
// MI455X (gfx1250) — matrix operand order repaired and checked by reading, not compiled, not run on hardware
//
#include <hip/hip_runtime.h>
#include <hip/hip_bf16.h>

// Problem constants (fixed by the reference)
#define B_  2
#define QL  2048
#define SL  2048
#define DM  1024
#define NH  16
#define AD  64
#define NEG_INF (-1e9f)

typedef __attribute__((ext_vector_type(16))) __bf16         v16bf;
typedef __attribute__((ext_vector_type(8)))  float          v8f;
typedef __attribute__((ext_vector_type(4)))  unsigned short v4u;
typedef __attribute__((ext_vector_type(8)))  unsigned short v8u;
typedef __attribute__((ext_vector_type(16))) unsigned short v16u;

// ---------------- fp32 -> bf16 (round to nearest even) ----------------
__device__ __forceinline__ unsigned short f2bf(float f) {
  unsigned u = __builtin_bit_cast(unsigned, f);
  u += 0x7FFFu + ((u >> 16) & 1u);
  return (unsigned short)(u >> 16);
}

// ---------------- WMMA fragment loaders (wave32) ----------------
// A-matrix 16x32 bf16: lane holds row M = lane&15.
// lanes 0-15 : elements 0..7 -> K=k0+0..7,   8..15 -> K=k0+16..23
// lanes 16-31: elements 0..7 -> K=k0+8..15,  8..15 -> K=k0+24..31
__device__ __forceinline__ v16bf load_frag_a(const unsigned short* base, int stride, int k0) {
  int lane = threadIdx.x & 31;
  const unsigned short* p = base + (long)(lane & 15) * stride + k0 + ((lane >> 4) << 3);
  v8u lo = *(const v8u*)p;
  v8u hi = *(const v8u*)(p + 16);
  v16u r = __builtin_shufflevector(lo, hi, 0,1,2,3,4,5,6,7,8,9,10,11,12,13,14,15);
  return __builtin_bit_cast(v16bf, r);
}

// B-matrix 32x16 bf16: lane holds column N = lane&15.
// lanes 0-15: K=k0+0..7, k0+16..23; lanes 16-31: K=k0+8..15, k0+24..31 (same as A).
// Source buffers are N-major (row n contiguous in K) -> two b128 loads.
__device__ __forceinline__ v16bf load_frag_b(const unsigned short* base, int stride, int k0) {
  int lane = threadIdx.x & 31;
  const unsigned short* p = base + (long)(lane & 15) * stride + k0 + ((lane >> 4) << 3);
  v8u lo = *(const v8u*)p;
  v8u hi = *(const v8u*)(p + 16);
  v16u r = __builtin_shufflevector(lo, hi, 0,1,2,3,4,5,6,7,8,9,10,11,12,13,14,15);
  return __builtin_bit_cast(v16bf, r);
}

__device__ __forceinline__ v8f wmma_bf16(v16bf a, v16bf b, v8f c) {
  return __builtin_amdgcn_wmma_f32_16x16x32_bf16(false, a, false, b, (short)0, c, false, false);
}

// ---------------- conversion kernels ----------------
__global__ void k_cvt(const float* __restrict__ src, unsigned short* __restrict__ dst, int n) {
  int i = (blockIdx.x * blockDim.x + threadIdx.x) * 4;
  if (i >= n) return;                  // all sizes are multiples of 4
  float4 f = *(const float4*)(src + i);
  v4u r; r.x = f2bf(f.x); r.y = f2bf(f.y); r.z = f2bf(f.z); r.w = f2bf(f.w);
  *(v4u*)(dst + i) = r;
}

// transpose-convert: src [R,C] f32 row-major -> dst [C,R] bf16 row-major
__global__ void k_cvtT(const float* __restrict__ src, unsigned short* __restrict__ dst, int R, int C) {
  long i = (long)blockIdx.x * blockDim.x + threadIdx.x;
  if (i >= (long)R * C) return;
  int r = (int)(i / C), c = (int)(i % C);
  dst[(long)c * R + r] = f2bf(src[i]);
}

// mask (int, nonzero => masked) -> additive f32 bias (0 / -1e9)
__global__ void k_bias(const int* __restrict__ m, float* __restrict__ bias, int n) {
  int i = blockIdx.x * blockDim.x + threadIdx.x;
  if (i < n) bias[i] = m[i] ? NEG_INF : 0.0f;
}

// ---------------- generic WMMA GEMM: out[M,N] = A[M,K] * Bt[N,K]^T ----------------
// wave tile 32(M) x 64(N); 8 waves/block. MODE selects output scatter:
//  0: f32 row-major [M,N]
//  1: Q-proj  -> bf16 [B,H,Q,A]
//  2: KV-proj -> K: bf16 [B,H,S,A] (out1), V: bf16 transposed [B,H,A,S] (out2)
template<int MODE>
__global__ void k_gemm(const unsigned short* __restrict__ A,
                       const unsigned short* __restrict__ Bt,
                       float* __restrict__ outF,
                       unsigned short* __restrict__ out1,
                       unsigned short* __restrict__ out2,
                       int M, int N, int K) {
  int wid = blockIdx.x * (blockDim.x >> 5) + (threadIdx.x >> 5);
  int tilesN = N >> 6;
  int tm = wid / tilesN, tn = wid % tilesN;
  int m0 = tm << 5, n0 = tn << 6;
  if (m0 >= M) return;

  v8f acc[2][4] = {};
  const unsigned short* a0p = A + (long)m0 * K;
  const unsigned short* a1p = A + (long)(m0 + 16) * K;

  for (int k = 0; k < K; k += 32) {
    v16bf a0 = load_frag_a(a0p, K, k);
    v16bf a1 = load_frag_a(a1p, K, k);
#pragma unroll
    for (int nt = 0; nt < 4; ++nt) {
      v16bf b = load_frag_b(Bt + (long)(n0 + nt * 16) * K, K, k);
      acc[0][nt] = wmma_bf16(a0, b, acc[0][nt]);
      acc[1][nt] = wmma_bf16(a1, b, acc[1][nt]);
    }
  }

  int lane = threadIdx.x & 31;
  int half = lane >> 4, nn = lane & 15;
#pragma unroll
  for (int mt = 0; mt < 2; ++mt)
#pragma unroll
    for (int nt = 0; nt < 4; ++nt)
#pragma unroll
      for (int v = 0; v < 8; ++v) {
        int m = m0 + mt * 16 + v + half * 8;
        int n = n0 + nt * 16 + nn;
        float val = acc[mt][nt][v];
        if (MODE == 0) {
          outF[(long)m * N + n] = val;
        } else if (MODE == 1) {
          int b = m >> 11, qi = m & (QL - 1);
          int h = n >> 6,  a  = n & (AD - 1);
          out1[(((long)(b * NH + h) * QL) + qi) * AD + a] = f2bf(val);
        } else {
          int b = m >> 11, s = m & (SL - 1);
          int which = n >> 10, h = (n >> 6) & (NH - 1), a = n & (AD - 1);
          if (which == 0)
            out1[(((long)(b * NH + h) * SL) + s) * AD + a] = f2bf(val);
          else
            out2[(((long)(b * NH + h) * AD) + a) * SL + s] = f2bf(val);
        }
      }
}

// ---------------- block-cooperative transposed flash attention ----------------
// One block (8 waves) = one (b,h) and 8 adjacent 16-query tiles.  Each 32-key
// chunk of K and V^T is staged into LDS ONCE per block and consumed by all 8
// waves (8x less global/L2 K/V traffic).  S^T = K * Q^T puts keys on the
// accumulator axis and queries on lanes: softmax max is in-lane VALU plus ONE
// xor-16 shuffle per chunk; the sum keeps per-half partials combined at the end.
// P^T staging is wave-private LDS: same-wave LDS ops are in-order -> no barrier.
__global__ void k_attn(const unsigned short* __restrict__ qb,   // [B,H,Q,A] bf16
                       const unsigned short* __restrict__ kb,   // [B,H,S,A] bf16
                       const unsigned short* __restrict__ vt,   // [B,H,A,S] bf16 (V^T)
                       const float* __restrict__ bias,          // [B,S] 0 / -1e9
                       unsigned short* __restrict__ ctx) {      // [B,Q,H*A] bf16
  __shared__ unsigned short ldsK[32 * 64];      // K chunk   [key][a]
  __shared__ unsigned short ldsV[64 * 32];      // V^T chunk [a][key-local]
  __shared__ unsigned short ldsP[8][16 * 32];   // per-wave P^T [query][key-local]
  int tid   = threadIdx.x;
  int wslot = tid >> 5, lane = tid & 31;
  int half  = lane >> 4, nn = lane & 15;
  int bh = blockIdx.x >> 4;                       // 16 blocks per (b,h)
  int q0 = ((blockIdx.x & 15) << 7) + (wslot << 4);
  int b = bh / NH, h = bh % NH;

  const unsigned short* qp  = qb + (((long)bh * QL) + q0) * AD;
  const unsigned short* kp  = kb + ((long)bh * SL) * AD;
  const unsigned short* vtp = vt + ((long)bh * AD) * SL;
  const float* bp = bias + b * SL;

  // cooperative staging indices: one b128 per thread per buffer
  int krow = tid >> 3, kcol = (tid & 7) << 3;     // K:   32 rows x 64
  int vrow = tid >> 2, vcol = (tid & 3) << 3;     // V^T: 64 rows x 32

  // Q tile as the B-operand of S^T = K * Q^T (held for the whole S loop)
  v16bf bq0 = load_frag_b(qp, AD, 0);
  v16bf bq1 = load_frag_b(qp, AD, 32);

  v8f acc[4] = {};          // ctx^T: rows a = t*16 + v + 8*half, col = query nn
  float mrow = -1e30f;      // per-query running max (uniform across the half pair)
  float lrow = 0.0f;        // per-query per-HALF partial sum (combined at end)

  for (int j0 = 0; j0 < SL; j0 += 32) {
    __syncthreads();        // previous chunk fully consumed before overwrite
    *(v8u*)&ldsK[krow * 64 + kcol] = *(const v8u*)(kp  + (long)(j0 + krow) * AD + kcol);
    *(v8u*)&ldsV[vrow * 32 + vcol] = *(const v8u*)(vtp + (long)vrow * SL + j0 + vcol);
    __syncthreads();        // staged K/V visible to all waves

    // ---- S^T tiles from LDS-resident K ----
    v8f s0 = {}, s1 = {};
    s0 = wmma_bf16(load_frag_a(ldsK, 64, 0),            bq0, s0);
    s0 = wmma_bf16(load_frag_a(ldsK, 64, 32),           bq1, s0);
    s1 = wmma_bf16(load_frag_a(ldsK + 16 * 64, 64, 0),  bq0, s1);
    s1 = wmma_bf16(load_frag_a(ldsK + 16 * 64, 64, 32), bq1, s1);

    // my lane's keys: j0 + jt*16 + 8*half + v (8 consecutive per tile)
    const float* bk0 = bp + j0 + 8 * half;
    const float* bk1 = bk0 + 16;

    float x0[8], x1[8];
    float xm = -1e30f;
#pragma unroll
    for (int v = 0; v < 8; ++v) { x0[v] = fmaf(s0[v], 0.125f, bk0[v]); xm = fmaxf(xm, x0[v]); }
#pragma unroll
    for (int v = 0; v < 8; ++v) { x1[v] = fmaf(s1[v], 0.125f, bk1[v]); xm = fmaxf(xm, x1[v]); }
    xm = fmaxf(xm, __shfl_xor(xm, 16, 32));     // full-row max before exp

    float mnew  = fmaxf(mrow, xm);
    float alpha = __expf(mrow - mnew);
    mrow = mnew;

    float ps = 0.0f;
    v8u pv0, pv1;
#pragma unroll
    for (int v = 0; v < 8; ++v) { float p = __expf(x0[v] - mnew); ps += p; pv0[v] = f2bf(p); }
#pragma unroll
    for (int v = 0; v < 8; ++v) { float p = __expf(x1[v] - mnew); ps += p; pv1[v] = f2bf(p); }
    lrow = lrow * alpha + ps;

#pragma unroll
    for (int t = 0; t < 4; ++t)
#pragma unroll
      for (int v = 0; v < 8; ++v) acc[t][v] *= alpha;   // alpha uniform per lane

    // ---- stage P^T [query][key] (two packed b128 DS stores), wave-private ----
    *(v8u*)&ldsP[wslot][nn * 32 + 8 * half]      = pv0;
    *(v8u*)&ldsP[wslot][nn * 32 + 16 + 8 * half] = pv1;
    v16bf bP = load_frag_b(&ldsP[wslot][0], 32, 0);

    // ---- PV from LDS-resident V^T ----
    acc[0] = wmma_bf16(load_frag_a(ldsV,           32, 0), bP, acc[0]);
    acc[1] = wmma_bf16(load_frag_a(ldsV + 16 * 32, 32, 0), bP, acc[1]);
    acc[2] = wmma_bf16(load_frag_a(ldsV + 32 * 32, 32, 0), bP, acc[2]);
    acc[3] = wmma_bf16(load_frag_a(ldsV + 48 * 32, 32, 0), bP, acc[3]);
  }

  // ---- combine per-half sums, normalize, pack, store ctx [B,Q,H*A] ----
  float l = lrow + __shfl_xor(lrow, 16, 32);
  float inv = 1.0f / l;
  unsigned short* cp = ctx + ((long)(b * QL + q0 + nn)) * DM + h * AD + 8 * half;
#pragma unroll
  for (int t = 0; t < 4; ++t) {
    v8u o;
#pragma unroll
    for (int v = 0; v < 8; ++v) o[v] = f2bf(acc[t][v] * inv);
    *(v8u*)(cp + t * 16) = o;
  }
}

// ---------------- host launch ----------------
extern "C" void kernel_launch(void* const* d_in, const int* in_sizes, int n_in,
                              void* d_out, int out_size, void* d_ws, size_t ws_size,
                              hipStream_t stream) {
  (void)in_sizes; (void)n_in; (void)out_size; (void)ws_size;
  const float* iQ  = (const float*)d_in[0];
  const float* iK  = (const float*)d_in[1];
  const int*   msk = (const int*)d_in[2];
  const float* Wq  = (const float*)d_in[3];
  const float* Wkv = (const float*)d_in[4];
  const float* Wo  = (const float*)d_in[5];
  float* out = (float*)d_out;

  unsigned short* ws = (unsigned short*)d_ws;
  size_t o = 0;
  unsigned short* iQb = ws + o; o += (size_t)B_ * QL * DM;       // bf16 iQ
  unsigned short* iKb = ws + o; o += (size_t)B_ * SL * DM;       // bf16 iK
  unsigned short* Wqt = ws + o; o += (size_t)DM * DM;            // Wq^T
  unsigned short* Wkt = ws + o; o += (size_t)2 * DM * DM;        // Wkv^T
  unsigned short* Wot = ws + o; o += (size_t)DM * DM;            // Wo^T
  unsigned short* qb  = ws + o; o += (size_t)B_ * NH * QL * AD;  // [B,H,Q,A]
  unsigned short* kb  = ws + o; o += (size_t)B_ * NH * SL * AD;  // [B,H,S,A]
  unsigned short* vt  = ws + o; o += (size_t)B_ * NH * AD * SL;  // [B,H,A,S]
  unsigned short* ctx = ws + o; o += (size_t)B_ * QL * DM;       // [B,Q,H*A]
  float* bias = (float*)(ws + o);                                // [B,S] f32

  // 1) precision conversion + weight transposes + mask bias
  k_cvt <<<(B_ * QL * DM) / (256 * 4), 256, 0, stream>>>(iQ, iQb, B_ * QL * DM);
  k_cvt <<<(B_ * SL * DM) / (256 * 4), 256, 0, stream>>>(iK, iKb, B_ * SL * DM);
  k_cvtT<<<(DM * DM) / 256,     256, 0, stream>>>(Wq,  Wqt, DM, DM);
  k_cvtT<<<(2 * DM * DM) / 256, 256, 0, stream>>>(Wkv, Wkt, DM, 2 * DM);
  k_cvtT<<<(DM * DM) / 256,     256, 0, stream>>>(Wo,  Wot, DM, DM);
  k_bias<<<(B_ * SL) / 256,     256, 0, stream>>>(msk, bias, B_ * SL);

  // 2) Q projection: [B*Q, D] @ [D, D] -> [B,H,Q,A]
  { int M = B_ * QL, N = DM, K = DM;
    int waves = (M / 32) * (N / 64);
    k_gemm<1><<<waves / 8, 256, 0, stream>>>(iQb, Wqt, nullptr, qb, nullptr, M, N, K); }

  // 3) fused KV projection: [B*S, D] @ [D, 2D] -> K [B,H,S,A], V^T [B,H,A,S]
  { int M = B_ * SL, N = 2 * DM, K = DM;
    int waves = (M / 32) * (N / 64);
    k_gemm<2><<<waves / 8, 256, 0, stream>>>(iKb, Wkt, nullptr, kb, vt, M, N, K); }

  // 4) masked flash attention (block-cooperative K/V staging) -> ctx
  { int blocks = B_ * NH * (QL / 128);   // 8 waves x 16 queries = 128 queries/block
    k_attn<<<blocks, 256, 0, stream>>>(qb, kb, vt, bias, ctx); }

  // 5) output projection -> f32 d_out
  { int M = B_ * QL, N = DM, K = DM;
    int waves = (M / 32) * (N / 64);
    k_gemm<0><<<waves / 8, 256, 0, stream>>>(ctx, Wot, out, nullptr, nullptr, M, N, K); }
}
